// MultiheadFlashDiff_6425271075516
// MI455X (gfx1250) — compile-verified, matrix operand order checked by reading, not run on hardware
//
#include <hip/hip_runtime.h>
#include <hip/hip_bf16.h>

// ---------------- problem constants ----------------
#define EMBED   1024
#define SEQ     2048
#define BATCH   2
#define NHEADS  8
#define NTOK    (BATCH*SEQ)          // 4096 tokens
#define KT      64                   // keys per attention iteration

static constexpr float LAMBDA_INIT   = 0.7836057665316244f;   // 0.8 - 0.6*exp(-3.6)
static constexpr float ONE_M_LAMBDA  = 1.0f - 0.7836057665316244f;
static constexpr float QSCALE        = 0.35355339059327373f;  // 64^(-1/4), folded into Q

// ---------------- WMMA types / helpers ----------------
typedef __bf16 v16bf __attribute__((ext_vector_type(16)));
typedef float  v8f   __attribute__((ext_vector_type(8)));

union Frag { v16bf v; uint4 q[2]; };

__device__ __forceinline__ v8f wmma_bf16(v16bf a, v16bf b, v8f c) {
    // D = A(16x32) * B(32x16) + C ; emits v_wmma_f32_16x16x32_bf16
    return __builtin_amdgcn_wmma_f32_16x16x32_bf16(
        false, a, false, b, (short)0, c, false, false);
}

__device__ __forceinline__ unsigned short f2bf(float f) {
    union { float f; unsigned u; } v; v.f = f;
    unsigned r = v.u + 0x7fffu + ((v.u >> 16) & 1u);   // round-to-nearest-even
    return (unsigned short)(r >> 16);
}
__device__ __forceinline__ float bf2f(unsigned short h) {
    union { unsigned u; float f; } v; v.u = ((unsigned)h) << 16;
    return v.f;
}

// ---------------- CDNA5 async global->LDS DMA (ASYNCcnt path) ----------------
// GLOBAL_LOAD_ASYNC_TO_LDS_B128: per-lane 16B copy memory->LDS with no VGPR staging.
// LDS address operand = wave-relative LDS byte offset = low 32 bits of flat shared ptr
// (ISA 10.2: LDS aperture address maps by truncation to addr[31:0]).
#define USE_ASYNC_COPY 1

#if USE_ASYNC_COPY
__device__ __forceinline__ void gld_async_b128(const unsigned short* g,
                                               const unsigned short* lds) {
    unsigned           loff = (unsigned)(uintptr_t)lds;
    unsigned long long ga   = (unsigned long long)(uintptr_t)g;
    asm volatile("global_load_async_to_lds_b128 %0, %1, off"
                 :: "v"(loff), "v"(ga) : "memory");
}
__device__ __forceinline__ void async_copy_done() {
    asm volatile("s_wait_asynccnt 0x0" ::: "memory");
}
#else
__device__ __forceinline__ void gld_async_b128(const unsigned short* g,
                                               const unsigned short* lds) {
    *reinterpret_cast<uint4*>(const_cast<unsigned short*>(lds)) =
        *reinterpret_cast<const uint4*>(g);
}
__device__ __forceinline__ void async_copy_done() {}
#endif

// ---- wave32 in-row (16-lane) butterfly exchange: VALU v_permlane16 if available ----
#if defined(__has_builtin)
#  if __has_builtin(__builtin_amdgcn_permlane16)
#    define USE_PERMLANE16 1
#  endif
#endif

template <int MASK>
__device__ __forceinline__ float bfly_xor(float x) {
#ifdef USE_PERMLANE16
    union { float f; int i; } u; u.f = x;
    int sel0, sel1;
    if constexpr (MASK == 1) { sel0 = (int)0x67452301u; sel1 = (int)0xEFCDAB89u; }
    else if constexpr (MASK == 2) { sel0 = (int)0x54761032u; sel1 = (int)0xDCFE98BAu; }
    else if constexpr (MASK == 4) { sel0 = (int)0x32107654u; sel1 = (int)0xBA98FEDCu; }
    else                          { sel0 = (int)0xFEDCBA98u; sel1 = (int)0x76543210u; }
    union { int i; float f; } r;
    r.i = __builtin_amdgcn_permlane16(u.i, u.i, sel0, sel1, false, false);
    return r.f;
#else
    return __shfl_xor(x, MASK, 32);
#endif
}
__device__ __forceinline__ float rowred_max(float t) {
    t = fmaxf(t, bfly_xor<1>(t));
    t = fmaxf(t, bfly_xor<2>(t));
    t = fmaxf(t, bfly_xor<4>(t));
    t = fmaxf(t, bfly_xor<8>(t));
    return t;
}
__device__ __forceinline__ float rowred_sum(float t) {
    t += bfly_xor<1>(t);
    t += bfly_xor<2>(t);
    t += bfly_xor<4>(t);
    t += bfly_xor<8>(t);
    return t;
}

// ---------------- elementwise kernels ----------------
__global__ void cvt_bf16_kernel(const float* __restrict__ src,
                                unsigned short* __restrict__ dst, int n) {
    int i = blockIdx.x * blockDim.x + threadIdx.x;
    if (i < n) dst[i] = f2bf(src[i]);
}

__global__ void lambda_kernel(const float* __restrict__ q1, const float* __restrict__ k1,
                              const float* __restrict__ q2, const float* __restrict__ k2,
                              float* __restrict__ lam) {
    if (threadIdx.x == 0) {
        float s1 = 0.f, s2 = 0.f;
        for (int i = 0; i < 64; ++i) { s1 += q1[i] * k1[i]; s2 += q2[i] * k2[i]; }
        lam[0] = __expf(s1) - __expf(s2) + LAMBDA_INIT;
    }
}

// RoPE (interleaved pairs) + rearrange [token][head*64+d] -> [(b*16+head)][s][64]; scale folds q-scaling.
__global__ void rope_kernel(const unsigned short* __restrict__ G,
                            unsigned short* __restrict__ O, float scale) {
    int idx = blockIdx.x * blockDim.x + threadIdx.x;      // NTOK*16*32 pairs
    if (idx >= NTOK * 16 * 32) return;
    int i     = idx & 31;
    int head  = (idx >> 5) & 15;
    int token = idx >> 9;
    int s = token & (SEQ - 1);
    int b = token >> 11;
    float x0 = bf2f(G[(size_t)token * 1024 + head * 64 + 2 * i]);
    float x1 = bf2f(G[(size_t)token * 1024 + head * 64 + 2 * i + 1]);
    float f  = (float)s * __powf(10000.f, -(float)(2 * i) / 64.f);
    float sn, c; __sincosf(f, &sn, &c);
    float y0 = (x0 * c - x1 * sn) * scale;
    float y1 = (x1 * c + x0 * sn) * scale;
    size_t o = ((size_t)(b * 16 + head) * SEQ + s) * 64 + 2 * i;
    O[o] = f2bf(y0); O[o + 1] = f2bf(y1);
}

// V rearrange + transpose: [token][h*128+dv] -> [(b*8+h)][dv][s]  (seq-contiguous rows)
__global__ void vrearr_kernel(const unsigned short* __restrict__ V,
                              unsigned short* __restrict__ O) {
    int idx = blockIdx.x * blockDim.x + threadIdx.x;      // NTOK*1024
    if (idx >= NTOK * 1024) return;
    int col = idx & 1023, token = idx >> 10;
    int s = token & (SEQ - 1), b = token >> 11;
    int h = col >> 7, dv = col & 127;
    O[((size_t)(b * 8 + h) * 128 + dv) * SEQ + s] = V[idx];
}

// ---------------- WMMA GEMM:  C[m][n] = sum_k A[m][k]*W[n][k]  (i.e. A @ W.T) ----------------
// Block = 128 threads = 4 waves; block tile 128x64; wave tile 32x64 (8 WMMAs / k-step).
template <bool OUT_BF16>
__global__ __launch_bounds__(128)
void gemm_xwT_kernel(const unsigned short* __restrict__ A,
                     const unsigned short* __restrict__ W,
                     void* __restrict__ Cout) {
    __shared__ __align__(16) unsigned short As[128 * 32];  // 8 KB
    __shared__ __align__(16) unsigned short Bs[64 * 32];   // 4 KB
    const int tid  = threadIdx.x;
    const int lane = tid & 31, wave = tid >> 5;
    const int hi   = lane >> 4, lp = lane & 15;
    const int m0   = blockIdx.x * 128;
    const int n0   = blockIdx.y * 64;

    v8f acc[2][4];
    v8f vz = {0.f,0.f,0.f,0.f,0.f,0.f,0.f,0.f};
#pragma unroll
    for (int mi = 0; mi < 2; ++mi)
#pragma unroll
        for (int n = 0; n < 4; ++n) acc[mi][n] = vz;

    const int browB = tid >> 1;
    const int bkoff = (tid & 1) * 16;

    for (int k0 = 0; k0 < 1024; k0 += 32) {
        // A tile: 128 rows x 32 halfs; one row (64 B) per thread -> 4 async b128
        {
            const unsigned short* ga = A + (size_t)(m0 + tid) * 1024 + k0;
#pragma unroll
            for (int i = 0; i < 4; ++i)
                gld_async_b128(ga + i * 8, &As[tid * 32 + i * 8]);
        }
        // B tile: 64 rows x 32 halfs; half-row (32 B) per thread -> 2 async b128
        {
            const unsigned short* gb = W + (size_t)(n0 + browB) * 1024 + k0 + bkoff;
            gld_async_b128(gb,     &Bs[browB * 32 + bkoff]);
            gld_async_b128(gb + 8, &Bs[browB * 32 + bkoff + 8]);
        }
        if (k0 + 32 < 1024) {
            __builtin_prefetch(A + (size_t)(m0 + tid) * 1024 + k0 + 32, 0, 2);
            __builtin_prefetch(W + (size_t)(n0 + browB) * 1024 + k0 + 32 + bkoff, 0, 2);
        }
        async_copy_done();
        __syncthreads();

        Frag aF[2];
#pragma unroll
        for (int mi = 0; mi < 2; ++mi) {
            const int row = wave * 32 + mi * 16 + lp;
            aF[mi].q[0] = *reinterpret_cast<const uint4*>(&As[row * 32 + hi * 8]);
            aF[mi].q[1] = *reinterpret_cast<const uint4*>(&As[row * 32 + 16 + hi * 8]);
        }
#pragma unroll
        for (int n = 0; n < 4; ++n) {
            Frag bF;
            bF.q[0] = *reinterpret_cast<const uint4*>(&Bs[(n * 16 + lp) * 32 + hi * 8]);
            bF.q[1] = *reinterpret_cast<const uint4*>(&Bs[(n * 16 + lp) * 32 + 16 + hi * 8]);
#pragma unroll
            for (int mi = 0; mi < 2; ++mi)
                acc[mi][n] = wmma_bf16(aF[mi].v, bF.v, acc[mi][n]);
        }
        __syncthreads();
    }
#pragma unroll
    for (int mi = 0; mi < 2; ++mi)
#pragma unroll
        for (int n = 0; n < 4; ++n)
#pragma unroll
            for (int r = 0; r < 8; ++r) {
                int row = m0 + wave * 32 + mi * 16 + r + 8 * hi;
                int col = n0 + n * 16 + lp;
                float v = acc[mi][n][r];
                if (OUT_BF16) ((unsigned short*)Cout)[(size_t)row * 1024 + col] = f2bf(v);
                else          ((float*)Cout)[(size_t)row * 1024 + col] = v;
            }
}

// ---------------- differential flash attention ----------------
// Block: 64 threads = 2 waves. Wave w = variant w (sub-head 2h+w) for one (b,h,16-query tile).
// 64-key iterations: 8 WMMAs QK^T + 16 WMMAs P.V; tiles staged with async global->LDS DMA.
__global__ __launch_bounds__(64)
void flashdiff_kernel(const unsigned short* __restrict__ Qb,
                      const unsigned short* __restrict__ Kb,
                      const unsigned short* __restrict__ Vt,
                      const float* __restrict__ lamp,
                      const float* __restrict__ lnw,
                      const float* __restrict__ lnb,
                      unsigned short* __restrict__ Ob) {
    __shared__ __align__(16) unsigned short Qs[2][16 * 64];   //  4 KB
    __shared__ __align__(16) unsigned short Ks[2][64 * 64];   // 16 KB
    __shared__ __align__(16) unsigned short Vts[128 * 64];    // 16 KB  [vdim][key]
    __shared__ __align__(16) unsigned short Ps[2][16 * 64];   //  4 KB
    __shared__ float Of[2][16 * 128];                         // 16 KB
    __shared__ float MsRed[16][4];

    const int tid  = threadIdx.x;
    const int lane = tid & 31, w = tid >> 5;
    const int hi   = lane >> 4, lp = lane & 15;

    const int qt = blockIdx.x & 127;
    const int h  = (blockIdx.x >> 7) & 7;
    const int b  = blockIdx.x >> 10;
    const int q0 = qt * 16;

    // --- Q tile (per-wave) via async DMA, then its two A fragments ---
    const size_t qbase = ((size_t)(b * 16 + 2 * h + w) * SEQ + q0) * 64;
    {
        int row = lane >> 1;
        int off = (lane & 1) * 32;
        const unsigned short* gq = Qb + qbase + (size_t)row * 64 + off;
#pragma unroll
        for (int i = 0; i < 4; ++i)
            gld_async_b128(gq + i * 8, &Qs[w][row * 64 + off + i * 8]);
        async_copy_done();
    }
    Frag qA[2];
#pragma unroll
    for (int c = 0; c < 2; ++c) {
        qA[c].q[0] = *reinterpret_cast<const uint4*>(&Qs[w][lp * 64 + c * 32 + hi * 8]);
        qA[c].q[1] = *reinterpret_cast<const uint4*>(&Qs[w][lp * 64 + c * 32 + 16 + hi * 8]);
    }

    const size_t kbase  = (size_t)(b * 16 + 2 * h + w) * SEQ * 64;
    const size_t vtbase = (size_t)(b * 8 + h) * 128 * SEQ;

    float mrow[8], lrow[8];
    v8f o[8];
    v8f vz = {0.f,0.f,0.f,0.f,0.f,0.f,0.f,0.f};
#pragma unroll
    for (int r = 0; r < 8; ++r) { mrow[r] = -1e30f; lrow[r] = 0.f; }
#pragma unroll
    for (int n = 0; n < 8; ++n) o[n] = vz;

    for (int it = 0; it < SEQ / KT; ++it) {
        const int kt0 = it * KT;
        __syncthreads();                       // previous shared V tile fully consumed

        // K tile (per-wave): 64 keys x 64 halfs; 2 key-rows per lane -> 16 async b128
#pragma unroll
        for (int rr = 0; rr < 2; ++rr) {
            const int key = lane * 2 + rr;
            const unsigned short* gk = Kb + kbase + (size_t)(kt0 + key) * 64;
#pragma unroll
            for (int i = 0; i < 8; ++i)
                gld_async_b128(gk + i * 8, &Ks[w][key * 64 + i * 8]);
        }
        // V tile (shared, [vdim][seq] in global): 128 vdim-rows x 64 keys
#pragma unroll
        for (int rr = 0; rr < 2; ++rr) {
            const int dv = tid * 2 + rr;
            const unsigned short* gv = Vt + vtbase + (size_t)dv * SEQ + kt0;
#pragma unroll
            for (int i = 0; i < 8; ++i)
                gld_async_b128(gv + i * 8, &Vts[dv * 64 + i * 8]);
        }
        if (it + 1 < SEQ / KT) {
            __builtin_prefetch(Kb + kbase + (size_t)(kt0 + KT + lane * 2) * 64, 0, 2);
            __builtin_prefetch(Vt + vtbase + (size_t)(tid * 2) * SEQ + kt0 + KT, 0, 2);
        }
        async_copy_done();
        __syncthreads();

        // ---- scores: S(16x64) = Q(16x64).K^T as 4 C tiles, 8 WMMAs ----
        v8f s[4];
#pragma unroll
        for (int t = 0; t < 4; ++t) s[t] = vz;
#pragma unroll
        for (int t = 0; t < 4; ++t)
#pragma unroll
            for (int c = 0; c < 2; ++c) {
                Frag kf;
                const int key = t * 16 + lp;
                kf.q[0] = *reinterpret_cast<const uint4*>(&Ks[w][key * 64 + c * 32 + hi * 8]);
                kf.q[1] = *reinterpret_cast<const uint4*>(&Ks[w][key * 64 + c * 32 + 16 + hi * 8]);
                s[t] = wmma_bf16(qA[c].v, kf.v, s[t]);
            }

        // ---- online softmax over 64 keys (row = r + 8*hi across a 16-lane group) ----
#pragma unroll
        for (int r = 0; r < 8; ++r) {
            float t01 = fmaxf(fmaxf(s[0][r], s[1][r]), fmaxf(s[2][r], s[3][r]));
            t01 = rowred_max(t01);
            float mn    = fmaxf(mrow[r], t01);
            float alpha = __expf(mrow[r] - mn);
            mrow[r] = mn;
            float p0 = __expf(s[0][r] - mn);
            float p1 = __expf(s[1][r] - mn);
            float p2 = __expf(s[2][r] - mn);
            float p3 = __expf(s[3][r] - mn);
            float rs = rowred_sum((p0 + p1) + (p2 + p3));
            lrow[r] = lrow[r] * alpha + rs;
#pragma unroll
            for (int n = 0; n < 8; ++n) o[n][r] = o[n][r] * alpha;
            const int prow = r + 8 * hi;
            Ps[w][prow * 64 +      lp] = f2bf(p0);
            Ps[w][prow * 64 + 16 + lp] = f2bf(p1);
            Ps[w][prow * 64 + 32 + lp] = f2bf(p2);
            Ps[w][prow * 64 + 48 + lp] = f2bf(p3);
        }

        // ---- O += P(16x64).V(64x128): 16 WMMAs ----
        Frag pA[2];
#pragma unroll
        for (int c = 0; c < 2; ++c) {
            pA[c].q[0] = *reinterpret_cast<const uint4*>(&Ps[w][lp * 64 + c * 32 + hi * 8]);
            pA[c].q[1] = *reinterpret_cast<const uint4*>(&Ps[w][lp * 64 + c * 32 + 16 + hi * 8]);
        }
#pragma unroll
        for (int n = 0; n < 8; ++n) {
            const int vd = n * 16 + lp;
#pragma unroll
            for (int c = 0; c < 2; ++c) {
                Frag vf;
                vf.q[0] = *reinterpret_cast<const uint4*>(&Vts[vd * 64 + c * 32 + hi * 8]);
                vf.q[1] = *reinterpret_cast<const uint4*>(&Vts[vd * 64 + c * 32 + 16 + hi * 8]);
                o[n] = wmma_bf16(pA[c].v, vf.v, o[n]);
            }
        }
    }

    // ---- per-variant normalized O -> LDS ----
#pragma unroll
    for (int r = 0; r < 8; ++r) {
        float inv = 1.f / lrow[r];
        int row = r + 8 * hi;
#pragma unroll
        for (int n = 0; n < 8; ++n)
            Of[w][row * 128 + n * 16 + lp] = o[n][r] * inv;
    }
    __syncthreads();

    // ---- combine o1 - lam*o2, RMS-norm, scale; all 64 threads (4 per row) ----
    {
        const float lam = lamp[0];
        const int row = tid >> 2;
        const int q4  = (tid & 3) * 32;
        float ms = 0.f;
        for (int dv = q4; dv < q4 + 32; ++dv) {
            float v = Of[0][row * 128 + dv] - lam * Of[1][row * 128 + dv];
            ms += v * v;
        }
        MsRed[row][tid & 3] = ms;
        __syncthreads();
        float tot = MsRed[row][0] + MsRed[row][1] + MsRed[row][2] + MsRed[row][3];
        float rinv = rsqrtf(tot * (1.f / 128.f) + 1e-8f);
        size_t obase = ((size_t)b * SEQ + q0 + row) * 1024 + h * 128;
        for (int dv = q4; dv < q4 + 32; ++dv) {
            float v = Of[0][row * 128 + dv] - lam * Of[1][row * 128 + dv];
            Ob[obase + dv] = f2bf((v * rinv * lnw[dv] + lnb[dv]) * ONE_M_LAMBDA);
        }
    }
}

// ---------------- host launcher ----------------
extern "C" void kernel_launch(void* const* d_in, const int* in_sizes, int n_in,
                              void* d_out, int out_size, void* d_ws, size_t ws_size,
                              hipStream_t stream) {
    (void)in_sizes; (void)n_in; (void)out_size; (void)ws_size;
    const float* x   = (const float*)d_in[0];
    const float* Wq  = (const float*)d_in[1];
    const float* Wk  = (const float*)d_in[2];
    const float* Wv  = (const float*)d_in[3];
    const float* Wo  = (const float*)d_in[4];
    const float* lq1 = (const float*)d_in[5];
    const float* lk1 = (const float*)d_in[6];
    const float* lq2 = (const float*)d_in[7];
    const float* lk2 = (const float*)d_in[8];
    const float* lnw = (const float*)d_in[9];
    const float* lnb = (const float*)d_in[10];

    char* ws = (char*)d_ws;
    size_t off = 0;
    float* lam = (float*)(ws + off);                    off += 256;
    unsigned short* xb   = (unsigned short*)(ws + off); off += (size_t)NTOK * 1024 * 2;   // 8 MB
    unsigned short* Wqb  = (unsigned short*)(ws + off); off += (size_t)1024 * 1024 * 2;   // 2 MB
    unsigned short* Wkb  = (unsigned short*)(ws + off); off += (size_t)1024 * 1024 * 2;
    unsigned short* Wvb  = (unsigned short*)(ws + off); off += (size_t)1024 * 1024 * 2;
    unsigned short* Wob  = (unsigned short*)(ws + off); off += (size_t)1024 * 1024 * 2;
    unsigned short* Qraw = (unsigned short*)(ws + off); off += (size_t)NTOK * 1024 * 2;   // 8 MB
    unsigned short* Kraw = (unsigned short*)(ws + off); off += (size_t)NTOK * 1024 * 2;
    unsigned short* Vraw = (unsigned short*)(ws + off); off += (size_t)NTOK * 1024 * 2;
    unsigned short* Qb   = (unsigned short*)(ws + off); off += (size_t)NTOK * 1024 * 2;
    unsigned short* Kb   = (unsigned short*)(ws + off); off += (size_t)NTOK * 1024 * 2;
    unsigned short* Vtb  = (unsigned short*)(ws + off); off += (size_t)NTOK * 1024 * 2;
    unsigned short* Ob   = Qraw;   // Qraw dead after rope_kernel; reuse for attention output

    const int T = 256;
    cvt_bf16_kernel<<<(NTOK * 1024 + T - 1) / T, T, 0, stream>>>(x,  xb,  NTOK * 1024);
    cvt_bf16_kernel<<<(1024 * 1024 + T - 1) / T, T, 0, stream>>>(Wq, Wqb, 1024 * 1024);
    cvt_bf16_kernel<<<(1024 * 1024 + T - 1) / T, T, 0, stream>>>(Wk, Wkb, 1024 * 1024);
    cvt_bf16_kernel<<<(1024 * 1024 + T - 1) / T, T, 0, stream>>>(Wv, Wvb, 1024 * 1024);
    cvt_bf16_kernel<<<(1024 * 1024 + T - 1) / T, T, 0, stream>>>(Wo, Wob, 1024 * 1024);
    lambda_kernel<<<1, 32, 0, stream>>>(lq1, lk1, lq2, lk2, lam);

    dim3 gg(NTOK / 128, 1024 / 64);
    gemm_xwT_kernel<true><<<gg, 128, 0, stream>>>(xb, Wqb, Qraw);
    gemm_xwT_kernel<true><<<gg, 128, 0, stream>>>(xb, Wkb, Kraw);
    gemm_xwT_kernel<true><<<gg, 128, 0, stream>>>(xb, Wvb, Vraw);

    const int npairs = NTOK * 16 * 32;
    rope_kernel<<<(npairs + T - 1) / T, T, 0, stream>>>(Qraw, Qb, QSCALE);
    rope_kernel<<<(npairs + T - 1) / T, T, 0, stream>>>(Kraw, Kb, 1.0f);
    vrearr_kernel<<<(NTOK * 1024 + T - 1) / T, T, 0, stream>>>(Vraw, Vtb);

    flashdiff_kernel<<<BATCH * NHEADS * (SEQ / 16), 64, 0, stream>>>(
        Qb, Kb, Vtb, lam, lnw, lnb, Ob);

    gemm_xwT_kernel<false><<<gg, 128, 0, stream>>>(Ob, Wob, (float*)d_out);
}
